// GIN_4647154614721
// MI455X (gfx1250) — matrix operand order repaired and checked by reading, not compiled, not run on hardware
//
#include <hip/hip_runtime.h>
#include <hip/hip_bf16.h>

// ---------- problem constants ----------
#define B_GR 4096
#define L_ND 56
#define N_ND (B_GR * L_ND)   // 229376

typedef __attribute__((ext_vector_type(16))) _Float16 v16h;
typedef __attribute__((ext_vector_type(8)))  _Float16 v8h;
typedef __attribute__((ext_vector_type(8)))  float    v8f;

static inline unsigned cdivu(long long a, long long b) { return (unsigned)((a + b - 1) / b); }

// ---------------------------------------------------------------- utilities
__global__ void k_zero_f32(float* __restrict__ p, long long n) {
    long long i = (long long)blockIdx.x * blockDim.x + threadIdx.x;
    if (i < n) p[i] = 0.f;
}
__global__ void k_zero_f16(_Float16* __restrict__ p, long long n) {
    long long i = (long long)blockIdx.x * blockDim.x + threadIdx.x;
    if (i < n) p[i] = (_Float16)0.f;
}

// edge scatter: agg[dst] += x[src]  (segment_sum over random edges)
__global__ void k_scatter(const float* __restrict__ x, const int* __restrict__ ei,
                          float* __restrict__ agg, int D, long long E) {
    long long i = (long long)blockIdx.x * blockDim.x + threadIdx.x;
    if (i >= E * (long long)D) return;
    long long e = i / D;
    int d = (int)(i - e * D);
    int s = ei[e];
    int t = ei[E + e];
    atomicAdd(&agg[(long long)t * D + d], x[(long long)s * D + d]);
}

// h16 = f16(x + agg)
__global__ void k_prep(const float* __restrict__ x, const float* __restrict__ agg,
                       _Float16* __restrict__ h, long long n) {
    long long i = (long long)blockIdx.x * blockDim.x + threadIdx.x;
    if (i < n) h[i] = (_Float16)(x[i] + agg[i]);
}

// transpose + f16 convert a [Din,Dout] f32 weight into Wt16 [Dout,Kp], K zero-padded
__global__ void k_wT(const float* __restrict__ src, _Float16* __restrict__ dst,
                     int Din, int Dout, int Kp) {
    long long i = (long long)blockIdx.x * blockDim.x + threadIdx.x;
    long long tot = (long long)Dout * Kp;
    if (i >= tot) return;
    int o = (int)(i / Kp);
    int k = (int)(i - (long long)o * Kp);
    dst[i] = (k < Din) ? (_Float16)src[(long long)k * Dout + o] : (_Float16)0.f;
}

// conv weight [O,C,Kt] (torch layout) -> Wt16 [O, Kt*C] with k = kk*C + c
__global__ void k_wconv(const float* __restrict__ src, _Float16* __restrict__ dst,
                        int O, int C, int Kt) {
    long long i = (long long)blockIdx.x * blockDim.x + threadIdx.x;
    long long tot = (long long)O * C * Kt;
    if (i >= tot) return;
    int o   = (int)(i / (C * Kt));
    int rem = (int)(i - (long long)o * C * Kt);
    int kk  = rem / C;
    int c   = rem - kk * C;
    dst[i] = (_Float16)src[((long long)o * C + c) * Kt + kk];
}

// -------------------------------------------------------------- WMMA GEMM
// D = act(A @ W + bias).  A: [M,Kp] f16 row-major (CONV=1: implicit im2col,
// row r -> &act[(b*Tin + t)*C], window of Kp contiguous f16).  Bt: [Nout,Kp] f16
// (pre-transposed, K padded).  One 16x16 tile per wave; 8 waves per block.
// ACT: 0 none, 1 ReLU, 2 ELU.
template <int CONV, int ACT>
__global__ __launch_bounds__(256) void k_gemm(
    const _Float16* __restrict__ A, const _Float16* __restrict__ Bt,
    const float* __restrict__ bias,
    _Float16* __restrict__ out16, float* __restrict__ out32,
    int M, int Kp, int Nout, int ldo16, int Tin, int Tout, int C) {

    int wave = threadIdx.x >> 5;
    long long tile = (long long)blockIdx.x * 8 + wave;
    int ntn = Nout >> 4;
    long long ntiles = (long long)(M >> 4) * ntn;
    if (tile >= ntiles) return;                  // wave-uniform: EXEC stays all-1s

    int rowTile = (int)(tile / ntn);
    int colTile = (int)(tile - (long long)rowTile * ntn);
    int lane = threadIdx.x & 31;
    int m  = lane & 15;                          // A row within tile == B column n
    int hh = lane >> 4;                          // half select
    int row = rowTile * 16 + m;
    int col = colTile * 16 + m;

    long long abase;
    if (CONV) {
        int b = row / Tout;
        int t = row - b * Tout;
        abase = ((long long)b * Tin + t) * C;    // contiguous Kp-window (channel-last)
    } else {
        abase = (long long)row * Kp;
    }
    const _Float16* arow = A + abase + 16 * hh;
    const _Float16* brow = Bt + (long long)col * Kp + 16 * hh;

    v8f acc = {};
    for (int k0 = 0; k0 < Kp; k0 += 32) {
        v16h av = *(const v16h*)(arow + k0);         // K = k0 + 16h .. +15, row m
        v16h bv = *(const v16h*)(brow + k0);         // K = k0 + 16h .. +15, col n
        acc = __builtin_amdgcn_wmma_f32_16x16x32_f16(
            false, av, false, bv, (short)0, acc, false, false);
    }

    float bb = bias[col];
#pragma unroll
    for (int r = 0; r < 8; ++r) {
        int orow = rowTile * 16 + r + 8 * hh;
        float v = acc[r] + bb;
        if (ACT == 1) v = fmaxf(v, 0.f);
        if (ACT == 2) v = (v > 0.f) ? v : (__expf(v) - 1.f);
        if (out16) out16[(long long)orow * ldo16 + col] = (_Float16)v;
        if (out32) out32[(long long)orow * Nout + col] = v;
    }
}

// avgpool2 over time, channel-last: dst[b,t,c] = .5*(src[b,2t,c]+src[b,2t+1,c])
__global__ void k_pool(const _Float16* __restrict__ src, _Float16* __restrict__ dst,
                       int Tin, int To, int C, long long tot) {
    long long i = (long long)blockIdx.x * blockDim.x + threadIdx.x;
    if (i >= tot) return;
    int c = (int)(i % C);
    long long bt = i / C;
    int t = (int)(bt % To);
    long long b = bt / To;
    long long s = ((long long)b * Tin + 2 * t) * C + c;
    dst[i] = (_Float16)(0.5f * ((float)src[s] + (float)src[s + C]));
}

// build fc1 input [B,96]: cols 0..79 = flatten of [B,16ch,5t] (c*5+t order) from
// p3 stored [B,5,16]; 80..87 = one_hot; 88..95 = zero pad (K -> 96)
__global__ void k_fcin(const _Float16* __restrict__ p3, const float* __restrict__ oh,
                       _Float16* __restrict__ fcin) {
    int i = blockIdx.x * blockDim.x + threadIdx.x;
    if (i >= B_GR * 96) return;
    int b = i / 96, j = i - b * 96;
    float v;
    if (j < 80) {
        int c = j / 5, t = j - c * 5;
        v = (float)p3[(long long)b * 80 + t * 16 + c];
    } else if (j < 88) {
        v = oh[b * 8 + (j - 80)];
    } else {
        v = 0.f;
    }
    fcin[i] = (_Float16)v;
}

// fc2: out[b] = sum_k a[b,k]*w[k] + bias   (block-per-row reduction, wave32)
__global__ __launch_bounds__(256) void k_fc2(const _Float16* __restrict__ a,
                                             const float* __restrict__ w,
                                             const float* __restrict__ bias,
                                             float* __restrict__ out, int K) {
    int b = blockIdx.x;
    float s = 0.f;
    for (int k = threadIdx.x; k < K; k += 256)
        s += (float)a[(long long)b * K + k] * w[k];
    for (int off = 16; off; off >>= 1) s += __shfl_xor(s, off, 32);
    __shared__ float red[8];
    int lane = threadIdx.x & 31, wv = threadIdx.x >> 5;
    if (lane == 0) red[wv] = s;
    __syncthreads();
    if (threadIdx.x == 0) {
        float t = 0.f;
        for (int i = 0; i < 8; ++i) t += red[i];
        out[b] = t + bias[0];
    }
}

// ---------------------------------------------------------------- host
extern "C" void kernel_launch(void* const* d_in, const int* in_sizes, int n_in,
                              void* d_out, int out_size, void* d_ws, size_t ws_size,
                              hipStream_t stream) {
    const float* x_in = (const float*)d_in[0];
    const int*   ei   = (const int*)d_in[1];
    const float* oh   = (const float*)d_in[3];
    const float* gw1[4], *gb1[4], *gw2[4], *gb2[4];
    for (int l = 0; l < 4; ++l) {
        gw1[l] = (const float*)d_in[4 + l * 4 + 0];
        gb1[l] = (const float*)d_in[4 + l * 4 + 1];
        gw2[l] = (const float*)d_in[4 + l * 4 + 2];
        gb2[l] = (const float*)d_in[4 + l * 4 + 3];
    }
    const float* c1w = (const float*)d_in[20]; const float* c1b = (const float*)d_in[21];
    const float* c2w = (const float*)d_in[22]; const float* c2b = (const float*)d_in[23];
    const float* c3w = (const float*)d_in[24]; const float* c3b = (const float*)d_in[25];
    const float* fc1w = (const float*)d_in[26]; const float* fc1b = (const float*)d_in[27];
    const float* fc2w = (const float*)d_in[28]; const float* fc2b = (const float*)d_in[29];
    const long long E = (long long)in_sizes[1] / 2;

    // ---- workspace layout (regions reused across stages) ----
    char* ws = (char*)d_ws;
    const size_t OFF_X   = 0;                       // N*128 f32 = 117.4 MB
    const size_t OFF_AGG = 117440512;               // N*128 f32
    const size_t OFF_H   = 234881024;               // N*128 f16 = 58.7 MB
    const size_t OFF_T   = 293601280;               // N*128 f16
    const size_t OFF_W   = 352321536;               // f16 weights ~1.7 MB

    float*    xbuf = (float*)(ws + OFF_X);
    float*    agg  = (float*)(ws + OFF_AGG);
    _Float16* h16  = (_Float16*)(ws + OFF_H);
    _Float16* t16  = (_Float16*)(ws + OFF_T);
    _Float16* wbase = (_Float16*)(ws + OFF_W);

    _Float16* w_g1a = wbase;            // 32*128
    _Float16* w_g1b = wbase + 4096;     // 128*128
    _Float16* w_g2a = wbase + 20480;    // 128*64
    _Float16* w_g2b = wbase + 28672;    // 64*64
    _Float16* w_g3a = wbase + 32768;    // 64*32
    _Float16* w_g3b = wbase + 34816;    // 32*32
    _Float16* w_g4a = wbase + 35840;    // 32*16
    _Float16* w_g4b = wbase + 36352;    // 16(pad32)*16
    _Float16* w_c1  = wbase + 36864;    // 64*160
    _Float16* w_c2  = wbase + 47104;    // 32*640
    _Float16* w_c3  = wbase + 67584;    // 16*320
    _Float16* w_fc1 = wbase + 72704;    // 8192*96

    // conv-stage aliases (safe: prior tenants dead before first write)
    _Float16* act0  = (_Float16*)(ws + OFF_H);                    // [B,56,16]
    _Float16* a1    = (_Float16*)(ws + OFF_X);                    // [B,47,64]
    _Float16* a2    = (_Float16*)(ws + OFF_T);                    // [B,38,32]
    _Float16* p2    = (_Float16*)(ws + OFF_T  + (16u << 20));     // [B,19,32]
    _Float16* a3    = (_Float16*)(ws + OFF_AGG);                  // [B,10,16]
    _Float16* p3    = (_Float16*)(ws + OFF_AGG + (2u << 20));     // [B,5,16]
    _Float16* fcin  = (_Float16*)(ws + OFF_AGG + (4u << 20));     // [B,96]
    _Float16* fc1o  = (_Float16*)(ws + OFF_AGG + (8u << 20));     // [B,8192]

    // ---- weight conversion (cheap, rerun each call for determinism) ----
    auto wt = [&](const float* s, _Float16* d, int Din, int Dout, int Kp) {
        long long n = (long long)Dout * Kp;
        k_wT<<<cdivu(n, 256), 256, 0, stream>>>(s, d, Din, Dout, Kp);
    };
    wt(gw1[0], w_g1a, 32, 128, 32);   wt(gw2[0], w_g1b, 128, 128, 128);
    wt(gw1[1], w_g2a, 128, 64, 128);  wt(gw2[1], w_g2b, 64, 64, 64);
    wt(gw1[2], w_g3a, 64, 32, 64);    wt(gw2[2], w_g3b, 32, 32, 32);
    wt(gw1[3], w_g4a, 32, 16, 32);    wt(gw2[3], w_g4b, 16, 16, 32);
    wt(fc1w,   w_fc1, 88, 8192, 96);
    k_wconv<<<cdivu(64 * 160, 256), 256, 0, stream>>>(c1w, w_c1, 64, 16, 10);
    k_wconv<<<cdivu(32 * 640, 256), 256, 0, stream>>>(c2w, w_c2, 32, 64, 10);
    k_wconv<<<cdivu(16 * 320, 256), 256, 0, stream>>>(c3w, w_c3, 16, 32, 10);

    // ---- GIN layers ----
    auto run_gin = [&](const float* xcur, int Din, int Dhid, int Dout,
                       const _Float16* wA, const float* bA,
                       const _Float16* wB, const float* bB,
                       float* xnext, _Float16* out16, int tKp) {
        long long nD = (long long)N_ND * Din;
        k_zero_f32<<<cdivu(nD, 256), 256, 0, stream>>>(agg, nD);
        k_scatter<<<cdivu(E * Din, 256), 256, 0, stream>>>(xcur, ei, agg, Din, E);
        k_prep<<<cdivu(nD, 256), 256, 0, stream>>>(xcur, agg, h16, nD);
        if (tKp != Dhid) {
            long long nt = (long long)N_ND * tKp;
            k_zero_f16<<<cdivu(nt, 256), 256, 0, stream>>>(t16, nt);
        }
        long long tA = (long long)(N_ND / 16) * (Dhid / 16);
        k_gemm<0, 1><<<cdivu(tA, 8), 256, 0, stream>>>(
            h16, wA, bA, t16, nullptr, N_ND, Din, Dhid, tKp, 1, 1, 1);
        long long tB = (long long)(N_ND / 16) * (Dout / 16);
        k_gemm<0, 2><<<cdivu(tB, 8), 256, 0, stream>>>(
            t16, wB, bB, out16, xnext, N_ND, tKp, Dout, Dout, 1, 1, 1);
    };
    run_gin(x_in, 32, 128, 128, w_g1a, gb1[0], w_g1b, gb2[0], xbuf, nullptr, 128);
    run_gin(xbuf, 128, 64, 64,  w_g2a, gb1[1], w_g2b, gb2[1], xbuf, nullptr, 64);
    run_gin(xbuf, 64, 32, 32,   w_g3a, gb1[2], w_g3b, gb2[2], xbuf, nullptr, 32);
    run_gin(xbuf, 32, 16, 16,   w_g4a, gb1[3], w_g4b, gb2[3], nullptr, act0, 32);

    // ---- convs as implicit-im2col WMMA GEMMs ----
    { // conv1: [B,56,16] -> relu -> [B,47,64]
        long long t = (long long)(B_GR * 47 / 16) * (64 / 16);
        k_gemm<1, 1><<<cdivu(t, 8), 256, 0, stream>>>(
            act0, w_c1, c1b, a1, nullptr, B_GR * 47, 160, 64, 64, 56, 47, 16);
    }
    { // conv2: [B,47,64] -> relu -> [B,38,32] -> pool -> [B,19,32]
        long long t = (long long)(B_GR * 38 / 16) * (32 / 16);
        k_gemm<1, 1><<<cdivu(t, 8), 256, 0, stream>>>(
            a1, w_c2, c2b, a2, nullptr, B_GR * 38, 640, 32, 32, 47, 38, 64);
        long long n = (long long)B_GR * 19 * 32;
        k_pool<<<cdivu(n, 256), 256, 0, stream>>>(a2, p2, 38, 19, 32, n);
    }
    { // conv3: [B,19,32] -> relu -> [B,10,16] -> pool -> [B,5,16]
        long long t = (long long)(B_GR * 10 / 16) * (16 / 16);
        k_gemm<1, 1><<<cdivu(t, 8), 256, 0, stream>>>(
            p2, w_c3, c3b, a3, nullptr, B_GR * 10, 320, 16, 16, 19, 10, 32);
        long long n = (long long)B_GR * 5 * 16;
        k_pool<<<cdivu(n, 256), 256, 0, stream>>>(a3, p3, 10, 5, 16, n);
    }

    // ---- head ----
    k_fcin<<<cdivu((long long)B_GR * 96, 256), 256, 0, stream>>>(p3, oh, fcin);
    { // fc1: [B,96(pad)] @ [96,8192] + b -> elu
        long long t = (long long)(B_GR / 16) * (8192 / 16);
        k_gemm<0, 2><<<cdivu(t, 8), 256, 0, stream>>>(
            fcin, w_fc1, fc1b, fc1o, nullptr, B_GR, 96, 8192, 8192, 1, 1, 1);
    }
    k_fc2<<<B_GR, 256, 0, stream>>>(fc1o, fc2w, fc2b, (float*)d_out, 8192);
}
